// StreamingMultiheadAttention_2714419331456
// MI455X (gfx1250) — matrix operand order repaired and checked by reading, not compiled, not run on hardware
//
#include <hip/hip_runtime.h>
#include <hip/hip_bf16.h>
#include <math.h>
#include <cstdint>

// Problem constants (from reference setup_inputs)
#define B_   8
#define T_   256
#define C_   1024
#define H_   16
#define D_   64
#define CAP_ 2048

typedef __bf16 bf16_t;
typedef __attribute__((ext_vector_type(16))) __bf16 v16bf;
typedef __attribute__((ext_vector_type(8)))  __bf16 v8bf;
typedef __attribute__((ext_vector_type(4)))  __bf16 v4bf;
typedef __attribute__((ext_vector_type(8)))  float  v8f;
typedef int v4i __attribute__((__vector_size__(16)));   // <4 x i32>

union V16U { v16bf v; v8bf h[2]; };

// ---------------------------------------------------------------------------
// CDNA5 async global->LDS path (ASYNCcnt-tracked DMA), with sync fallback.
// ---------------------------------------------------------------------------
#if defined(__has_builtin)
#if __has_builtin(__builtin_amdgcn_global_load_async_to_lds_b128) && \
    __has_builtin(__builtin_amdgcn_s_wait_asynccnt)
#define HAVE_ASYNC_LDS 1
#endif
#endif
#ifndef HAVE_ASYNC_LDS
#define HAVE_ASYNC_LDS 0
#endif

#define AS1 __attribute__((address_space(1)))
#define AS3 __attribute__((address_space(3)))

__device__ __forceinline__ void async_copy16(const bf16_t* g, bf16_t* l) {
#if HAVE_ASYNC_LDS
  // Builtin expects typed int4 pointers: AS1 source, AS3 (LDS) destination.
  // Generic shared pointers carry the LDS byte offset in their low 32 bits.
  __builtin_amdgcn_global_load_async_to_lds_b128(
      (AS1 v4i*)(uintptr_t)g,
      (AS3 v4i*)(unsigned)(uintptr_t)l, 0, 0);
#else
  *reinterpret_cast<v8bf*>(l) = *reinterpret_cast<const v8bf*>(g);
#endif
}

__device__ __forceinline__ void wait_async0() {
#if HAVE_ASYNC_LDS
  __builtin_amdgcn_s_wait_asynccnt(0);
#endif
}

// ---------------------------------------------------------------------------
// WMMA helper: D = A(16x32 bf16) * B(32x16 bf16) + C(16x16 f32)
// ---------------------------------------------------------------------------
__device__ __forceinline__ v8f wmma_bf16(v16bf a, v16bf b, v8f c) {
  return __builtin_amdgcn_wmma_f32_16x16x32_bf16(
      false, a, false, b, (short)0, c, false, false);
}

// A fragment (16x32 MxK) from row-major [M][K], leading dim ld.
// lane L: row M=L%16; K chunks {0..7,16..23} (L<16) / {8..15,24..31} (L>=16).
__device__ __forceinline__ v16bf load_a_frag(const bf16_t* tile, int ld) {
  const int lane = threadIdx.x & 31;
  const bf16_t* p = tile + (size_t)(lane & 15) * ld + ((lane & 16) ? 8 : 0);
  V16U u;
  u.h[0] = *reinterpret_cast<const v8bf*>(p);
  u.h[1] = *reinterpret_cast<const v8bf*>(p + 16);
  return u.v;
}

// B fragment (32x16 KxN) from [N][K] storage (row n contiguous in k).
// lane L: col N=L%16; K chunks {0..7,16..23} (L<16) / {8..15,24..31} (L>=16).
__device__ __forceinline__ v16bf load_b_frag(const bf16_t* tileT, int ld) {
  const int lane = threadIdx.x & 31;
  const bf16_t* p = tileT + (size_t)(lane & 15) * ld + ((lane & 16) ? 8 : 0);
  V16U u;
  u.h[0] = *reinterpret_cast<const v8bf*>(p);
  u.h[1] = *reinterpret_cast<const v8bf*>(p + 16);
  return u.v;
}

// ---------------------------------------------------------------------------
// fp32 -> bf16 conversion (4 elems/thread, vectorized)
// ---------------------------------------------------------------------------
__global__ void cvt_f32_bf16(const float* __restrict__ src,
                             bf16_t* __restrict__ dst, int n) {
  int i = (blockIdx.x * blockDim.x + threadIdx.x) * 4;
  if (i + 3 < n) {
    float4 f = *reinterpret_cast<const float4*>(src + i);
    v4bf o;
    o[0] = (bf16_t)f.x; o[1] = (bf16_t)f.y; o[2] = (bf16_t)f.z; o[3] = (bf16_t)f.w;
    *reinterpret_cast<v4bf*>(dst + i) = o;
  }
}

// V cache: fp32 [BH, CAP, D] -> bf16 transposed [BH, D, CAP] (write-coalesced)
__global__ void cvt_vcache_t(const float* __restrict__ src,
                             bf16_t* __restrict__ dst) {
  size_t i = (size_t)blockIdx.x * blockDim.x + threadIdx.x;
  int    s  = (int)(i % CAP_);
  size_t r  = i / CAP_;
  int    d  = (int)(r % D_);
  size_t bh = r / D_;
  dst[i] = (bf16_t)src[(bh * CAP_ + s) * D_ + d];
}

// ---------------------------------------------------------------------------
// In-projection GEMM (register-blocked 64x64 per wave) + RoPE + ring scatter.
// proj[m,n] = sum_k qin[m,k] * w[n,k];  M=B*T=2048, N=3C=3072, K=C=1024.
// ---------------------------------------------------------------------------
__global__ __launch_bounds__(32)
void inproj_rope_scatter(const bf16_t* __restrict__ qin,   // [B*T, C]
                         const bf16_t* __restrict__ w,     // [3C, C]
                         const int*    __restrict__ offset_p,
                         bf16_t* __restrict__ q_rope,      // [B,H,T,D]
                         bf16_t* __restrict__ k_cache,     // [B,H,CAP,D]
                         bf16_t* __restrict__ v_cache) {   // [B,H,D,CAP]
  const int ntn  = (3 * C_) / 64;            // 48 column blocks
  const int m0   = (blockIdx.x / ntn) * 64;
  const int n0   = (blockIdx.x % ntn) * 64;
  const int lane = threadIdx.x & 31;
  const int mh   = (lane & 16) ? 8 : 0;

  v8f acc[4][4] = {};
  for (int k = 0; k < C_; k += 32) {
    v16bf a[4], b[4];
#pragma unroll
    for (int i = 0; i < 4; ++i)
      a[i] = load_a_frag(qin + (size_t)(m0 + 16 * i) * C_ + k, C_);
#pragma unroll
    for (int j = 0; j < 4; ++j)
      b[j] = load_b_frag(w + (size_t)(n0 + 16 * j) * C_ + k, C_);
#pragma unroll
    for (int i = 0; i < 4; ++i)
#pragma unroll
      for (int j = 0; j < 4; ++j)
        acc[i][j] = wmma_bf16(a[i], b[j], acc[i][j]);
  }

  const int offset = *offset_p;
  const float nfreq = -0.28782313662425572f;   // -ln(10000)*2/D

#pragma unroll
  for (int j = 0; j < 4; ++j) {
    const int n    = n0 + 16 * j + (lane & 15);
    const int sect = n / C_;                   // 0=q 1=k 2=v
    const int cn   = n % C_;
    const int head = cn / D_;
    const int d    = cn % D_;
    const float fr = __expf(nfreq * (float)(d >> 1));
#pragma unroll
    for (int i = 0; i < 4; ++i) {
#pragma unroll
      for (int r = 0; r < 8; ++r) {
        const int m = m0 + 16 * i + r + mh;
        const int bidx = m / T_;
        const int t    = m % T_;
        float val = acc[i][j][r];
        float partner = __shfl_xor(val, 1, 32);   // RoPE pair: adjacent lane
        if (sect < 2) {
          float ang = (float)(offset + t) * fr;
          float c = __cosf(ang), s = __sinf(ang);
          float y = (d & 1) ? (partner * s + val * c)
                            : (val * c - partner * s);
          if (sect == 0) {
            q_rope[(((size_t)bidx * H_ + head) * T_ + t) * D_ + d] = (bf16_t)y;
          } else {
            int slot = (offset + t) % CAP_;
            k_cache[(((size_t)bidx * H_ + head) * CAP_ + slot) * D_ + d] =
                (bf16_t)y;
          }
        } else {
          int slot = (offset + t) % CAP_;
          v_cache[(((size_t)bidx * H_ + head) * D_ + d) * CAP_ + slot] =
              (bf16_t)val;
        }
      }
    }
  }
}

// ---------------------------------------------------------------------------
// Flash attention: 4-wave workgroup covers (b, h, 64 t-rows).  K/V chunks
// (32 cache slots) staged once into double-buffered LDS via the async
// global->LDS DMA and shared by all waves; each wave owns a 16-row t tile.
// ---------------------------------------------------------------------------
__global__ __launch_bounds__(128)
void attention(const bf16_t* __restrict__ q_rope,   // [B,H,T,D]
               const bf16_t* __restrict__ k_cache,  // [B,H,CAP,D]
               const bf16_t* __restrict__ v_cache,  // [B,H,D,CAP]
               const int*    __restrict__ offset_p,
               bf16_t* __restrict__ attn_out) {     // [B,T,C]
  __shared__ bf16_t Kc[2][32 * D_];   // [s][d]  4KB per buffer
  __shared__ bf16_t Vc[2][D_ * 32];   // [d][s]  4KB per buffer
  __shared__ bf16_t Pl[4][16 * 32];   // per-wave P staging

  const int wgs  = T_ / 64;                      // 4 workgroups per (b,h)
  const int bh   = blockIdx.x / wgs;
  const int bidx = bh / H_;
  const int head = bh % H_;
  const int tid  = threadIdx.x;
  const int wid  = tid >> 5;
  const int lane = tid & 31;
  const int mh   = (lane & 16) ? 8 : 0;
  const int t0   = (blockIdx.x % wgs) * 64 + wid * 16;

  const int offset  = *offset_p;
  const int last    = offset + T_ - 1;
  const int end_idx = last % CAP_;

  const bf16_t* qbase = q_rope + ((size_t)bh * T_ + t0) * D_;
  const bf16_t* kbase = k_cache + (size_t)bh * CAP_ * D_;
  const bf16_t* vbase = v_cache + (size_t)bh * D_ * CAP_;

  const v16bf aq0 = load_a_frag(qbase + 0,  D_);
  const v16bf aq1 = load_a_frag(qbase + 32, D_);

  v8f o0 = {}, o1 = {}, o2 = {}, o3 = {};
  float rmax[8], rsum[8];
#pragma unroll
  for (int r = 0; r < 8; ++r) { rmax[r] = -3.0e38f; rsum[r] = 0.f; }

  const float scale = 0.125f;                    // 1/sqrt(64)
  const int nchunks = CAP_ / 32;                 // 64

  // Cooperative chunk staging: K chunk is 4KB contiguous; V chunk is 64 rows
  // of 64B from the transposed cache.  4 x b128 async copies per thread.
  auto stage = [&](int s0, int buf) {
    const bf16_t* gk = kbase + (size_t)s0 * D_ + tid * 8;
    async_copy16(gk,        &Kc[buf][tid * 8]);
    async_copy16(gk + 1024, &Kc[buf][tid * 8 + 1024]);
#pragma unroll
    for (int q = 0; q < 2; ++q) {
      int p = tid + q * 128;
      int row = p >> 2, off = (p & 3) * 8;
      async_copy16(vbase + (size_t)row * CAP_ + s0 + off,
                   &Vc[buf][row * 32 + off]);
    }
  };

  stage(0, 0);
  wait_async0();
  __syncthreads();

  for (int c = 0; c < nchunks; ++c) {
    const int s0 = c * 32;
    if (c + 1 < nchunks) stage(s0 + 32, (c + 1) & 1);

    const bf16_t* kbuf = Kc[c & 1];
    const bf16_t* vbuf = Vc[c & 1];

    // ---- scores: Q(16x64) x K^T(64x32) as 4 WMMAs from LDS operands
    v8f sc[2];
#pragma unroll
    for (int nt = 0; nt < 2; ++nt) {
      v16bf bk0 = load_b_frag(kbuf + (nt * 16) * D_ + 0,  D_);
      v16bf bk1 = load_b_frag(kbuf + (nt * 16) * D_ + 32, D_);
      v8f a = {};
      a = wmma_bf16(aq0, bk0, a);
      a = wmma_bf16(aq1, bk1, a);
      sc[nt] = a;
    }

    // ---- ring-buffer position recovery + sliding-window causal mask
#pragma unroll
    for (int nt = 0; nt < 2; ++nt) {
      int s     = s0 + nt * 16 + (lane & 15);
      int delta = s - end_idx;
      int pk    = (delta <= 0) ? (last + delta) : (last + delta - CAP_);
      if (s >= offset + T_) pk = -1;
#pragma unroll
      for (int r = 0; r < 8; ++r) {
        int pq = offset + t0 + r + mh;
        int dd = pq - pk;
        bool valid = (pk >= 0) && (dd >= 0) && (dd < CAP_);
        sc[nt][r] = valid ? sc[nt][r] * scale : -3.0e38f;
      }
    }

    // ---- online softmax (row stats live in half-wave lane groups)
    float cf[8];
#pragma unroll
    for (int r = 0; r < 8; ++r) {
      float mx = fmaxf(sc[0][r], sc[1][r]);
#pragma unroll
      for (int w = 1; w < 16; w <<= 1) mx = fmaxf(mx, __shfl_xor(mx, w, 32));
      float nm = fmaxf(rmax[r], mx);
      cf[r] = __expf(rmax[r] - nm);
      rmax[r] = nm;
    }
#pragma unroll
    for (int r = 0; r < 8; ++r) {
      o0[r] *= cf[r]; o1[r] *= cf[r]; o2[r] *= cf[r]; o3[r] *= cf[r];
      float p0 = (sc[0][r] > -1.0e37f) ? __expf(sc[0][r] - rmax[r]) : 0.f;
      float p1 = (sc[1][r] > -1.0e37f) ? __expf(sc[1][r] - rmax[r]) : 0.f;
      float ss = p0 + p1;
#pragma unroll
      for (int w = 1; w < 16; w <<= 1) ss += __shfl_xor(ss, w, 32);
      rsum[r] = rsum[r] * cf[r] + ss;
      int mrow = r + mh;
      Pl[wid][mrow * 32 + (lane & 15)]      = (bf16_t)p0;
      Pl[wid][mrow * 32 + 16 + (lane & 15)] = (bf16_t)p1;
    }
    __syncthreads();   // P store->load fence (also keeps waves in lockstep)

    // ---- P(16x32) x V(32x64): A from LDS, B from transposed V in LDS
    {
      const bf16_t* p = &Pl[wid][0] + (size_t)(lane & 15) * 32 +
                        ((lane & 16) ? 8 : 0);
      V16U u;
      u.h[0] = *reinterpret_cast<const v8bf*>(p);
      u.h[1] = *reinterpret_cast<const v8bf*>(p + 16);
      v16bf ap = u.v;
      o0 = wmma_bf16(ap, load_b_frag(vbuf + 0 * 32 * 16, 32), o0);
      o1 = wmma_bf16(ap, load_b_frag(vbuf + 1 * 32 * 16, 32), o1);
      o2 = wmma_bf16(ap, load_b_frag(vbuf + 2 * 32 * 16, 32), o2);
      o3 = wmma_bf16(ap, load_b_frag(vbuf + 3 * 32 * 16, 32), o3);
    }

    wait_async0();     // next chunk's DMA has landed
    __syncthreads();   // all waves done with this buffer before overwrite
  }

  // ---- normalize, emit [B,T,C] bf16 for the out-projection
#pragma unroll
  for (int r = 0; r < 8; ++r) {
    float inv = 1.0f / rsum[r];
    int t = t0 + r + mh;
    size_t base = ((size_t)bidx * T_ + t) * C_ + head * D_;
    int dl = lane & 15;
    attn_out[base + 0  + dl] = (bf16_t)(o0[r] * inv);
    attn_out[base + 16 + dl] = (bf16_t)(o1[r] * inv);
    attn_out[base + 32 + dl] = (bf16_t)(o2[r] * inv);
    attn_out[base + 48 + dl] = (bf16_t)(o3[r] * inv);
  }
}

// ---------------------------------------------------------------------------
// Out-projection GEMM (register-blocked 64x64 per wave), f32 result.
// ---------------------------------------------------------------------------
__global__ __launch_bounds__(32)
void outproj(const bf16_t* __restrict__ x,   // [B*T, C]
             const bf16_t* __restrict__ w,   // [C, C]
             float* __restrict__ out) {      // [B*T, C]
  const int ntn  = C_ / 64;                  // 16
  const int m0   = (blockIdx.x / ntn) * 64;
  const int n0   = (blockIdx.x % ntn) * 64;
  const int lane = threadIdx.x & 31;
  const int mh   = (lane & 16) ? 8 : 0;

  v8f acc[4][4] = {};
  for (int k = 0; k < C_; k += 32) {
    v16bf a[4], b[4];
#pragma unroll
    for (int i = 0; i < 4; ++i)
      a[i] = load_a_frag(x + (size_t)(m0 + 16 * i) * C_ + k, C_);
#pragma unroll
    for (int j = 0; j < 4; ++j)
      b[j] = load_b_frag(w + (size_t)(n0 + 16 * j) * C_ + k, C_);
#pragma unroll
    for (int i = 0; i < 4; ++i)
#pragma unroll
      for (int j = 0; j < 4; ++j)
        acc[i][j] = wmma_bf16(a[i], b[j], acc[i][j]);
  }
#pragma unroll
  for (int i = 0; i < 4; ++i)
#pragma unroll
    for (int j = 0; j < 4; ++j) {
      const int n = n0 + 16 * j + (lane & 15);
#pragma unroll
      for (int r = 0; r < 8; ++r)
        out[(size_t)(m0 + 16 * i + r + mh) * C_ + n] = acc[i][j][r];
    }
}

// ---------------------------------------------------------------------------
extern "C" void kernel_launch(void* const* d_in, const int* in_sizes, int n_in,
                              void* d_out, int out_size, void* d_ws, size_t ws_size,
                              hipStream_t stream) {
  const float* query = (const float*)d_in[0];            // [B,T,C]
  const float* kv    = (const float*)d_in[1];            // [2,B,H,CAP,D]
  const float* w_in  = (const float*)d_in[2];            // [3C,C]
  const float* w_out = (const float*)d_in[3];            // [C,C]
  const int*   offp  = (const int*)d_in[4];
  float*       out   = (float*)d_out;

  const size_t nQ  = (size_t)B_ * T_ * C_;
  const size_t nWI = (size_t)3 * C_ * C_;
  const size_t nWO = (size_t)C_ * C_;
  const size_t nKC = (size_t)B_ * H_ * CAP_ * D_;

  char* ws = (char*)d_ws;
  size_t off = 0;
  auto alloc = [&](size_t bytes) -> void* {
    void* p = ws + off;
    off += (bytes + 255) & ~(size_t)255;
    return p;
  };
  bf16_t* q_in_bf  = (bf16_t*)alloc(nQ  * 2);
  bf16_t* w_in_bf  = (bf16_t*)alloc(nWI * 2);
  bf16_t* w_out_bf = (bf16_t*)alloc(nWO * 2);
  bf16_t* q_rope   = (bf16_t*)alloc(nQ  * 2);            // [B,H,T,D]
  bf16_t* k_bf     = (bf16_t*)alloc(nKC * 2);            // [B,H,CAP,D]
  bf16_t* v_bf     = (bf16_t*)alloc(nKC * 2);            // [B,H,D,CAP]
  bf16_t* attn_bf  = (bf16_t*)alloc(nQ  * 2);            // [B,T,C]

  // 1) precision conversion of activations, weights and KV cache
  cvt_f32_bf16<<<(int)(nQ  / 1024), 256, 0, stream>>>(query, q_in_bf, (int)nQ);
  cvt_f32_bf16<<<(int)(nWI / 1024), 256, 0, stream>>>(w_in,  w_in_bf, (int)nWI);
  cvt_f32_bf16<<<(int)(nWO / 1024), 256, 0, stream>>>(w_out, w_out_bf, (int)nWO);
  cvt_f32_bf16<<<(int)(nKC / 1024), 256, 0, stream>>>(kv, k_bf, (int)nKC);
  cvt_vcache_t<<<(int)(nKC / 256), 256, 0, stream>>>(kv + nKC, v_bf);

  // 2) in-projection + RoPE + ring scatter (M=2048, N=3072, K=1024)
  inproj_rope_scatter<<<(B_ * T_ / 64) * (3 * C_ / 64), 32, 0, stream>>>(
      q_in_bf, w_in_bf, offp, q_rope, k_bf, v_bf);

  // 3) flash attention, async-LDS double-buffered KV streaming
  attention<<<B_ * H_ * (T_ / 64), 128, 0, stream>>>(
      q_rope, k_bf, v_bf, offp, attn_bf);

  // 4) out-projection (M=2048, N=1024, K=1024) -> f32 output
  outproj<<<(B_ * T_ / 64) * (C_ / 64), 32, 0, stream>>>(
      attn_bf, w_out_bf, out);
}
